// BahdanauAttention_72945724555371
// MI455X (gfx1250) — matrix operand order repaired and checked by reading, not compiled, not run on hardware
//
#include <hip/hip_runtime.h>
#include <hip/hip_bf16.h>

// ---------------------------------------------------------------------------
// Bahdanau attention, MI455X (gfx1250), wave32 + bf16 WMMA.
// B=64, T=2048, H=512, A=512.
// ---------------------------------------------------------------------------

#define B_ 64
#define T_ 2048
#define H_ 512
#define A_ 512
#define LDSK 520   // padded bf16 row stride (1040 B) -> conflict-free ds_load_b128

typedef __attribute__((ext_vector_type(16))) __bf16 v16bf;
typedef __attribute__((ext_vector_type(8)))  __bf16 v8bf;
typedef __attribute__((ext_vector_type(8)))  float  v8f;

__device__ __forceinline__ unsigned pk2bf(float a, float b) {
    union { __bf16 h[2]; unsigned u; } t;
    t.h[0] = (__bf16)a; t.h[1] = (__bf16)b;
    return t.u;
}

// ---------------------------------------------------------------------------
// Prep 1: W_enc fp32 [A,H] -> bf16 (row-major, contiguous K per row)
// ---------------------------------------------------------------------------
__global__ void __launch_bounds__(256)
cvt_wenc_kernel(const float* __restrict__ W, unsigned short* __restrict__ Wb) {
    int i = blockIdx.x * 256 + threadIdx.x;          // float4 index, 65536 total
    float4 f = ((const float4*)W)[i];
    uint2 o;
    o.x = pk2bf(f.x, f.y);
    o.y = pk2bf(f.z, f.w);
    ((uint2*)Wb)[i] = o;
}

// ---------------------------------------------------------------------------
// Prep 2: dec_proj[b,a] = sum_h decoder_hidden[b,h] * W_dec[a,h]   (fp32)
// ---------------------------------------------------------------------------
__global__ void __launch_bounds__(256)
dec_proj_kernel(const float* __restrict__ dh, const float* __restrict__ Wd,
                float* __restrict__ out) {
    int idx = blockIdx.x * 256 + threadIdx.x;        // 32768 = B*A
    int b = idx >> 9;
    int a = idx & (A_ - 1);
    const float4* x = (const float4*)(dh + (size_t)b * H_);
    const float4* w = (const float4*)(Wd + (size_t)a * H_);
    float acc = 0.f;
#pragma unroll 4
    for (int i = 0; i < H_ / 4; ++i) {
        float4 p = x[i], q = w[i];
        acc += p.x * q.x + p.y * q.y + p.z * q.z + p.w * q.w;
    }
    out[idx] = acc;
}

// ---------------------------------------------------------------------------
// Main: scores[b,t] = sum_a V[a] * tanh( (Enc W_enc^T)[b,t,a] + dec_proj[b,a] )
// One WG per (b, 16-row t tile). 8 waves; wave w owns A-columns [64w, 64w+64).
// GEMM via v_wmma_f32_16x16x32_bf16, K=512 in 16 steps.
// ---------------------------------------------------------------------------
__global__ void __launch_bounds__(256)
bahdanau_scores_kernel(const float* __restrict__ enc,
                       const unsigned short* __restrict__ Wbf,
                       const float* __restrict__ decProj,
                       const float* __restrict__ Vvec,
                       float* __restrict__ scores) {
    __shared__ __align__(16) unsigned short sA[16 * LDSK];  // 16x512 bf16 tile, padded
    __shared__ float sScore[16];

    const int tid  = threadIdx.x;
    const int wave = tid >> 5;
    const int lane = tid & 31;
    const int b    = blockIdx.x >> 7;                // 128 t-tiles per batch
    const int t0   = (blockIdx.x & 127) << 4;

    if (tid < 16) sScore[tid] = 0.0f;

    // ---- stage enc tile [16 x 512] fp32 -> bf16 LDS (each element converted once)
    const float* encTile = enc + ((size_t)b * T_ + t0) * H_;
    for (int i = tid; i < 16 * 128; i += 256) {      // 128 float4 per row
        int r  = i >> 7;
        int c4 = i & 127;
        float4 f = ((const float4*)(encTile + (size_t)r * H_))[c4];
        uint2 o;
        o.x = pk2bf(f.x, f.y);
        o.y = pk2bf(f.z, f.w);
        *(uint2*)&sA[r * LDSK + c4 * 4] = o;
    }
    __syncthreads();

    const int mrow   = lane & 15;
    const int hiHalf = lane >> 4;                    // lane half selects K sub-chunks
    const unsigned short* sArow = sA + mrow * LDSK;
    const unsigned short* wBase = Wbf + (size_t)(wave * 64 + mrow) * H_;

    v8f acc[4] = {};                                 // 4 N-tiles of 16 columns

    for (int k = 0; k < H_; k += 32) {
        // A fragment: lanes 0-15: K k..k+15, lanes 16-31: K k+16..k+31
        int kA = k + (hiHalf ? 16 : 0);
        v8bf alo = *(const v8bf*)(sArow + kA);
        v8bf ahi = *(const v8bf*)(sArow + kA + 8);
        v16bf afrag = __builtin_shufflevector(alo, ahi,
            0, 1, 2, 3, 4, 5, 6, 7, 8, 9, 10, 11, 12, 13, 14, 15);

        // B fragments: lanes 0-15 hold K k..k+15, lanes 16-31 hold K k+16..k+31
        int kB = k + (hiHalf ? 16 : 0);
#pragma unroll
        for (int j = 0; j < 4; ++j) {
            v16bf bfrag = *(const v16bf*)(wBase + (size_t)j * 16 * H_ + kB);
            acc[j] = __builtin_amdgcn_wmma_f32_16x16x32_bf16(
                false, afrag, false, bfrag, (short)0, acc[j], false, false);
        }
    }

    // ---- epilogue: + dec_proj, tanh, * V[a], reduce over A
    // C layout: lane holds column a_j = wave*64 + j*16 + mrow,
    //           rows M = i + (hiHalf ? 8 : 0) for VGPR i.
    float s[8] = {0.f, 0.f, 0.f, 0.f, 0.f, 0.f, 0.f, 0.f};
#pragma unroll
    for (int j = 0; j < 4; ++j) {
        int aj = wave * 64 + j * 16 + mrow;
        float d  = decProj[b * A_ + aj];
        float vj = Vvec[aj];
#pragma unroll
        for (int i = 0; i < 8; ++i)
            s[i] += vj * tanhf(acc[j][i] + d);
    }

    // reduce over the 16 columns held by each lane half (wave32 shuffles)
#pragma unroll
    for (int i = 0; i < 8; ++i) {
        float v = s[i];
        v += __shfl_xor(v, 1);
        v += __shfl_xor(v, 2);
        v += __shfl_xor(v, 4);
        v += __shfl_xor(v, 8);
        s[i] = v;
    }
    if ((lane & 15) == 0) {
        int mBase = hiHalf ? 8 : 0;
#pragma unroll
        for (int i = 0; i < 8; ++i)
            atomicAdd(&sScore[mBase + i], s[i]);     // ds_add_f32 across 8 waves
    }
    __syncthreads();

    if (tid < 16)
        scores[(size_t)b * T_ + t0 + tid] = sScore[tid];
}

// ---------------------------------------------------------------------------
// Softmax over T per batch row, in place. One WG per b, 8 elems per thread.
// ---------------------------------------------------------------------------
__global__ void __launch_bounds__(256)
softmax_rows_kernel(float* __restrict__ buf) {
    __shared__ float red[8];
    const int b = blockIdx.x, tid = threadIdx.x;
    const int wave = tid >> 5, lane = tid & 31;
    float* row = buf + (size_t)b * T_;

    float loc[8];
    float m = -3.402823466e38f;
#pragma unroll
    for (int q = 0; q < 8; ++q) {
        loc[q] = row[tid + q * 256];
        m = fmaxf(m, loc[q]);
    }
#pragma unroll
    for (int d = 16; d >= 1; d >>= 1) m = fmaxf(m, __shfl_xor(m, d));
    if (lane == 0) red[wave] = m;
    __syncthreads();
    float M = red[0];
#pragma unroll
    for (int i = 1; i < 8; ++i) M = fmaxf(M, red[i]);
    __syncthreads();

    float s = 0.f;
#pragma unroll
    for (int q = 0; q < 8; ++q) {
        float e = __expf(loc[q] - M);
        loc[q] = e;
        s += e;
    }
#pragma unroll
    for (int d = 16; d >= 1; d >>= 1) s += __shfl_xor(s, d);
    if (lane == 0) red[wave] = s;
    __syncthreads();
    float S = 0.f;
#pragma unroll
    for (int i = 0; i < 8; ++i) S += red[i];
    float inv = 1.0f / S;
#pragma unroll
    for (int q = 0; q < 8; ++q) row[tid + q * 256] = loc[q] * inv;
}

// ---------------------------------------------------------------------------
// context[b,h] = sum_t w[b,t] * enc[b,t,h]. One WG per b; 2 h-cols per thread.
// ---------------------------------------------------------------------------
__global__ void __launch_bounds__(256)
context_kernel(const float* __restrict__ enc, const float* __restrict__ w,
               float* __restrict__ out) {
    const int b = blockIdx.x, tid = threadIdx.x;
    const float* e  = enc + (size_t)b * T_ * H_;
    const float* wr = w + (size_t)b * T_;
    float a0 = 0.f, a1 = 0.f;
#pragma unroll 4
    for (int t = 0; t < T_; ++t) {
        float wt = wr[t];                            // uniform -> scalar load
        a0 = fmaf(wt, e[(size_t)t * H_ + tid],       a0);
        a1 = fmaf(wt, e[(size_t)t * H_ + tid + 256], a1);
    }
    out[(size_t)b * H_ + tid]       = a0;
    out[(size_t)b * H_ + tid + 256] = a1;
}

// ---------------------------------------------------------------------------
// Launch. Inputs: encoder_outputs, decoder_hidden, W_enc, W_dec, V (all fp32).
// d_out = [context 64x512 | attention_weights 64x2048] fp32.
// Workspace: dec_proj (128 KB) + W_enc bf16 (512 KB) = 640 KB.
// ---------------------------------------------------------------------------
extern "C" void kernel_launch(void* const* d_in, const int* in_sizes, int n_in,
                              void* d_out, int out_size, void* d_ws, size_t ws_size,
                              hipStream_t stream) {
    const float* enc  = (const float*)d_in[0];
    const float* dech = (const float*)d_in[1];
    const float* Wenc = (const float*)d_in[2];
    const float* Wdec = (const float*)d_in[3];
    const float* V    = (const float*)d_in[4];

    float* ctx     = (float*)d_out;
    float* weights = ctx + B_ * H_;                  // scores -> softmax in place

    float*          decProj = (float*)d_ws;
    unsigned short* Wbf     = (unsigned short*)((char*)d_ws + (size_t)B_ * A_ * sizeof(float));

    cvt_wenc_kernel<<<(A_ * H_ / 4) / 256, 256, 0, stream>>>(Wenc, Wbf);
    dec_proj_kernel<<<(B_ * A_) / 256, 256, 0, stream>>>(dech, Wdec, decProj);
    bahdanau_scores_kernel<<<B_ * (T_ / 16), 256, 0, stream>>>(enc, Wbf, decProj, V, weights);
    softmax_rows_kernel<<<B_, 256, 0, stream>>>(weights);
    context_kernel<<<B_, 256, 0, stream>>>(enc, weights, ctx);
}
